// Qunatile_Network_4758823764520
// MI455X (gfx1250) — matrix operand order repaired and checked by reading, not compiled, not run on hardware
//
#include <hip/hip_runtime.h>
#include <hip/hip_bf16.h>

// ---------------------------------------------------------------------------
// Quantile network for MI455X (gfx1250, wave32, WMMA, async-LDS copies).
//
// Restructured math:
//   base = data @ W0[1:,:]                       (WMMA bf16 GEMM, 1.07 GFLOP)
//   h0[n,m] = relu(base[n] + t_k[m]*W0[0,:]+b0)  (broadcast, fused)
//   h1 = relu(h0 @ W1 + b1)                      (WMMA bf16 GEMM, 137 GFLOP)  <- hot
//   logit = h1 @ W2 + b2  -> softplus            (fused epilogue)
//   DCT-II over 64 nodes, Chebyshev coeffs, Clenshaw per quantile.
//
// Hot kernel: A (h0 tile) register-resident; B (W1 tile) staged per-workgroup
// into LDS via GLOBAL_LOAD_ASYNC_TO_LDS_B128 (ASYNCcnt), double-buffered.
//
// Workspace layout (needs ~11.8 MB):
//   [0)          dataB  bf16 [4096][256]         2,097,152 B
//   [2097152)    W0B    bf16 [512 cols][256 k]     262,144 B  (col-major)
//   [2359296)    W1B    bf16 [512 cols][512 k]     524,288 B  (col-major)
//   [2883584)    base   f32  [4096][512]         8,388,608 B
//   [11272192)   out2   f32  [4096][64]          1,048,576 B
// ---------------------------------------------------------------------------

typedef __attribute__((ext_vector_type(16))) __bf16 v16bf;
typedef __attribute__((ext_vector_type(8)))  __bf16 v8bf;
typedef __attribute__((ext_vector_type(8)))  float  v8f;

#define PI_F 3.14159265358979323846f

static __device__ __forceinline__ unsigned short f2bf_bits(float f) {
  unsigned u = __float_as_uint(f);
  u += 0x7fffu + ((u >> 16) & 1u);     // round-to-nearest-even
  return (unsigned short)(u >> 16);
}
static __device__ __forceinline__ __bf16 f2bf(float f) {
  unsigned short s = f2bf_bits(f);
  __bf16 r;
  __builtin_memcpy(&r, &s, 2);
  return r;
}
static __device__ __forceinline__ v8f wmma_bf16(v16bf a, v16bf b, v8f c) {
  // (neg_a, A, neg_b, B, c_mod, C, reuse_a, reuse_b)
  return __builtin_amdgcn_wmma_f32_16x16x32_bf16(false, a, false, b,
                                                 (short)0, c, false, false);
}

// --- gfx1250 async global->LDS copy (ASYNCcnt-tracked), inline asm ---------
static __device__ __forceinline__ void async_b128(unsigned lds_addr,
                                                  unsigned goff,
                                                  const void* sbase) {
  asm volatile("global_load_async_to_lds_b128 %0, %1, %2"
               :: "v"(lds_addr), "v"(goff), "s"(sbase)
               : "memory");
}
static __device__ __forceinline__ void wait_async_le4() {
  asm volatile("s_wait_asynccnt 0x4" ::: "memory");
}
static __device__ __forceinline__ void wait_async_le0() {
  asm volatile("s_wait_asynccnt 0x0" ::: "memory");
}

// ---------------------------------------------------------------------------
// Kernel 1: fp32 -> bf16 conversions (data row-major; W0[1:], W1 col-major).
// ---------------------------------------------------------------------------
__global__ void qn_convert(const float* __restrict__ data,
                           const float* __restrict__ W0,
                           const float* __restrict__ W1,
                           unsigned short* __restrict__ dataB,
                           unsigned short* __restrict__ W0B,
                           unsigned short* __restrict__ W1B) {
  const int ND = 4096 * 256, NW0 = 512 * 256, NW1 = 512 * 512;
  const int total = ND + NW0 + NW1;
  for (int idx = blockIdx.x * blockDim.x + threadIdx.x; idx < total;
       idx += gridDim.x * blockDim.x) {
    if (idx < ND) {
      dataB[idx] = f2bf_bits(data[idx]);
    } else if (idx < ND + NW0) {
      int t = idx - ND;
      int c = t >> 8, k = t & 255;                 // W0B[c][k] = W0[1+k][c]
      W0B[t] = f2bf_bits(W0[(1 + k) * 512 + c]);
    } else {
      int t = idx - ND - NW0;
      int c = t >> 9, k = t & 511;                 // W1B[c][k] = W1[k][c]
      W1B[t] = f2bf_bits(W1[k * 512 + c]);
    }
  }
}

// ---------------------------------------------------------------------------
// Kernel 2: base = dataB[4096x256] @ W0B (bf16, f32 accum). One 16x16 tile
// per wave; 8 k-steps split over two accumulator chains for WMMA ILP.
// ---------------------------------------------------------------------------
__global__ __launch_bounds__(256) void qn_base_gemm(
    const unsigned short* __restrict__ dataB,
    const unsigned short* __restrict__ W0B,
    float* __restrict__ baseOut) {
  const int lane = threadIdx.x & 31;
  const int tile = blockIdx.x * 8 + (threadIdx.x >> 5);  // 8192 tiles
  const int mtile = tile >> 5;
  const int nt = tile & 31;
  const int arow = mtile * 16 + (lane & 15);
  const int aoff = (lane >= 16) ? 16 : 0;
  const int col = nt * 16 + (lane & 15);
  const int boff = (lane >= 16) ? 16 : 0;

  v8f acc0 = {0.f, 0.f, 0.f, 0.f, 0.f, 0.f, 0.f, 0.f};
  v8f acc1 = {0.f, 0.f, 0.f, 0.f, 0.f, 0.f, 0.f, 0.f};
#pragma unroll
  for (int kk = 0; kk < 8; kk += 2) {
#pragma unroll
    for (int u = 0; u < 2; ++u) {
      v8bf a0 = *reinterpret_cast<const v8bf*>(dataB + arow * 256 + (kk + u) * 32 + aoff);
      v8bf a1 = *reinterpret_cast<const v8bf*>(dataB + arow * 256 + (kk + u) * 32 + aoff + 8);
      v16bf a;
#pragma unroll
      for (int j = 0; j < 8; ++j) { a[j] = a0[j]; a[j + 8] = a1[j]; }
      v16bf b = *reinterpret_cast<const v16bf*>(W0B + col * 256 + (kk + u) * 32 + boff);
      if (u == 0) acc0 = wmma_bf16(a, b, acc0);
      else        acc1 = wmma_bf16(a, b, acc1);
    }
  }
#pragma unroll
  for (int vr = 0; vr < 8; ++vr) {
    int Mr = mtile * 16 + vr + ((lane >= 16) ? 8 : 0);
    baseOut[Mr * 512 + nt * 16 + (lane & 15)] = acc0[vr] + acc1[vr];
  }
}

// ---------------------------------------------------------------------------
// Kernel 3 (hot): fused  h0 -> h1 -> logit -> softplus.
// Rows r = n*64 + m, 262144 total. Each wave owns 16 rows; h0 tile (16x512)
// lives in register-resident A fragments (16 x v16bf, 128 VGPR) and is reused
// across all 32 N-tiles of W1 (512 WMMA/wave). The 16KB W1 N-tile is shared
// by the whole workgroup: staged into LDS with async b128 copies, double-
// buffered (prefetch next tile while current one feeds the WMMAs).
// ---------------------------------------------------------------------------
#define BT_STRIDE 520                       // halves per LDS row (pad vs 512)

__global__ __launch_bounds__(256) void qn_mlp(
    const float* __restrict__ baseA,   // [4096][512]
    const float* __restrict__ W0,      // row 0 used (t-column weights)
    const float* __restrict__ b0,
    const unsigned short* __restrict__ W1B,  // [512 cols][512 k] bf16
    const float* __restrict__ b1v,
    const float* __restrict__ W2,      // [512]
    const float* __restrict__ b2v,
    float* __restrict__ out2) {        // [4096*64]
  __shared__ unsigned short Btile[2][16 * BT_STRIDE];   // 2 x 16.25 KB

  const int tid = threadIdx.x;
  const int lane = tid & 31;
  const int wave = tid >> 5;
  const int rbase = blockIdx.x * 128 + wave * 16;   // rows rbase..rbase+15
  const int n = rbase >> 6;                         // same sample for all 16 rows
  const int m = (rbase & 63) + (lane & 15);         // node index for this A row
  const float tkm = 0.5f * cosf(PI_F * (m + 0.5f) / 64.0f) + 0.5f;
  const int aoff = (lane >= 16) ? 8 : 0;

  // ---- build h0 tile in A-fragment register layout -------------------------
  v16bf afrag[16];
#pragma unroll
  for (int kk = 0; kk < 16; ++kk) {
#pragma unroll
    for (int half = 0; half < 2; ++half) {
      const int cb = kk * 32 + aoff + half * 16;    // 8 consecutive hidden units
      const float4* bp = reinterpret_cast<const float4*>(baseA + n * 512 + cb);
      const float4* wp = reinterpret_cast<const float4*>(W0 + cb);
      const float4* zp = reinterpret_cast<const float4*>(b0 + cb);
      float4 bv0 = bp[0], bv1 = bp[1];
      float4 wv0 = wp[0], wv1 = wp[1];
      float4 zv0 = zp[0], zv1 = zp[1];
      float v[8] = {bv0.x + tkm * wv0.x + zv0.x, bv0.y + tkm * wv0.y + zv0.y,
                    bv0.z + tkm * wv0.z + zv0.z, bv0.w + tkm * wv0.w + zv0.w,
                    bv1.x + tkm * wv1.x + zv1.x, bv1.y + tkm * wv1.y + zv1.y,
                    bv1.z + tkm * wv1.z + zv1.z, bv1.w + tkm * wv1.w + zv1.w};
#pragma unroll
      for (int j = 0; j < 8; ++j) {
        float h = v[j] > 0.f ? v[j] : 0.f;          // relu (layer 0)
        afrag[kk][half * 8 + j] = f2bf(h);
      }
    }
  }

  // ---- async staging helper: 16KB tile (16 cols x 512 k), 4 b128 per lane --
  const int crow = tid >> 4;                        // 0..15 (tile column)
  const int cseg = tid & 15;                        // 0..15 (64B segment)
  const unsigned lds0 =
      (unsigned)(size_t)(&Btile[0][0]) + (unsigned)(crow * (BT_STRIDE * 2) + cseg * 64);
  const unsigned lds1 =
      (unsigned)(size_t)(&Btile[1][0]) + (unsigned)(crow * (BT_STRIDE * 2) + cseg * 64);
  const unsigned gseg = (unsigned)(crow * 1024 + cseg * 64);   // within 16KB tile

  // prologue: stage tile 0 into buffer 0
  {
    const void* gsrc = (const void*)(W1B + 0 * 16 * 512);
#pragma unroll
    for (int q = 0; q < 4; ++q) async_b128(lds0 + q * 16u, gseg + q * 16u, gsrc);
  }

  // ---- layer 1 GEMM + fused layer 2 dot ------------------------------------
  float lsum[8] = {0.f, 0.f, 0.f, 0.f, 0.f, 0.f, 0.f, 0.f};
  const int boff = (lane >= 16) ? 8 : 0;
  const int lrow = lane & 15;

  for (int nt = 0; nt < 32; ++nt) {
    if (nt + 1 < 32) {                    // prefetch next tile into other buffer
      const void* gsrc = (const void*)(W1B + (nt + 1) * 16 * 512);
      const unsigned ldsn = ((nt + 1) & 1) ? lds1 : lds0;
#pragma unroll
      for (int q = 0; q < 4; ++q) async_b128(ldsn + q * 16u, gseg + q * 16u, gsrc);
      wait_async_le4();                   // oldest 4 (current tile) complete
    } else {
      wait_async_le0();
    }
    __syncthreads();                      // whole tile visible to all waves

    const unsigned short* Bt = &Btile[nt & 1][0];
    const int col = nt * 16 + lrow;
    v8f acc0 = {0.f, 0.f, 0.f, 0.f, 0.f, 0.f, 0.f, 0.f};
    v8f acc1 = {0.f, 0.f, 0.f, 0.f, 0.f, 0.f, 0.f, 0.f};
#pragma unroll
    for (int kk = 0; kk < 16; kk += 2) {
#pragma unroll
      for (int u = 0; u < 2; ++u) {
        const unsigned short* bp = Bt + lrow * BT_STRIDE + (kk + u) * 32 + boff;
        v8bf blo = *reinterpret_cast<const v8bf*>(bp);
        v8bf bhi = *reinterpret_cast<const v8bf*>(bp + 16);
        v16bf bf;
#pragma unroll
        for (int j = 0; j < 8; ++j) { bf[j] = blo[j]; bf[j + 8] = bhi[j]; }
        if (u == 0) acc0 = wmma_bf16(afrag[kk], bf, acc0);
        else        acc1 = wmma_bf16(afrag[kk + 1], bf, acc1);
      }
    }
    const float bc = b1v[col];
    const float wc = W2[col];
#pragma unroll
    for (int vr = 0; vr < 8; ++vr) {
      float h1 = acc0[vr] + acc1[vr] + bc;
      h1 = h1 > 0.f ? h1 : 0.f;                     // relu (layer 1)
      lsum[vr] += h1 * wc;                          // layer 2 partial dot
    }
    __syncthreads();                      // done reading buf before overwrite
  }

  // ---- reduce over the 16 columns held per lane-half, then activation ------
#pragma unroll
  for (int vr = 0; vr < 8; ++vr) {
#pragma unroll
    for (int off = 1; off < 16; off <<= 1)
      lsum[vr] += __shfl_xor(lsum[vr], off, 32);
  }
  if ((lane & 15) == 0) {
    const int rowb = (lane >= 16) ? 8 : 0;
    const float bb2 = b2v[0];
#pragma unroll
    for (int vr = 0; vr < 8; ++vr) {
      float x = lsum[vr] + bb2 + 1e-5f;             // logit + 1e-5
      float sp = (x > 20.f) ? x : log1pf(expf(x));  // softplus
      out2[rbase + rowb + vr] = sp + 0.001f;
    }
  }
}

// ---------------------------------------------------------------------------
// Kernel 4: periodic reindex + DCT-II + Chebyshev coeffs + C0 + Clenshaw.
// One 64-thread block per sample.
// ---------------------------------------------------------------------------
__global__ __launch_bounds__(64) void qn_final(const float* __restrict__ out2,
                                               const float* __restrict__ pred,
                                               float* __restrict__ out) {
  __shared__ float full[64];
  __shared__ float c[64];
  __shared__ float C[64];
  __shared__ float C0s;
  const int n = blockIdx.x;
  const int t = threadIdx.x;

  full[t] = out2[n * 64 + ((t + n) & 63)];          // full_output[n, t]
  __syncthreads();

  float s = 0.f;                                    // DCT-II, norm=None
  for (int j = 0; j < 64; ++j)
    s += full[j] * cosf(PI_F * (2.f * j + 1.f) * (float)t / 128.f);
  c[t] = 2.f * s;
  __syncthreads();

  float Ct;
  if (t == 0)        Ct = 0.f;
  else if (t <= 62)  Ct = (c[t - 1] - c[t + 1]) / (4.f * (float)t);
  else               Ct = c[62] * 0.25f * 63.f;
  C[t] = Ct;
  __syncthreads();

  if (t == 0) {
    float acc = 0.f;
    for (int e = 2; e < 64; e += 2) {
      float sg = ((e >> 1) & 1) ? -1.f : 1.f;
      acc += C[e] * sg;
    }
    C0s = 2.f * pred[n] - 2.f * acc;
  }
  __syncthreads();

  if (t < 7) {
    const float Q[7] = {0.05f, 0.1f, 0.25f, 0.5f, 0.75f, 0.9f, 0.95f};
    const float sigma = 2.f * Q[t] - 1.f;
    float T1 = 0.f, T2 = 0.f;
    for (int i = 63; i >= 1; --i) {                 // Clenshaw
      float nT = 2.f * sigma * T1 - T2 + C[i];
      T2 = T1;
      T1 = nT;
    }
    out[n * 7 + t] = sigma * T1 - T2 + 0.5f * C0s;
  }
}

// ---------------------------------------------------------------------------
extern "C" void kernel_launch(void* const* d_in, const int* in_sizes, int n_in,
                              void* d_out, int out_size, void* d_ws,
                              size_t ws_size, hipStream_t stream) {
  const float* data = (const float*)d_in[0];   // [4096][256]
  const float* pred = (const float*)d_in[1];   // [4096]
  const float* W0   = (const float*)d_in[2];   // [257][512]
  const float* b0   = (const float*)d_in[3];   // [512]
  const float* W1   = (const float*)d_in[4];   // [512][512]
  const float* b1   = (const float*)d_in[5];   // [512]
  const float* W2   = (const float*)d_in[6];   // [512]
  const float* b2   = (const float*)d_in[7];   // [1]

  char* ws = (char*)d_ws;
  unsigned short* dataB = (unsigned short*)(ws + 0);
  unsigned short* W0B   = (unsigned short*)(ws + 2097152);
  unsigned short* W1B   = (unsigned short*)(ws + 2359296);
  float*          baseA = (float*)(ws + 2883584);
  float*          out2  = (float*)(ws + 11272192);

  qn_convert<<<5632, 256, 0, stream>>>(data, W0, W1, dataB, W0B, W1B);
  qn_base_gemm<<<1024, 256, 0, stream>>>(dataB, W0B, baseA);
  qn_mlp<<<2048, 256, 0, stream>>>(baseA, W0, b0, W1B, b1, W2, b2, out2);
  qn_final<<<4096, 64, 0, stream>>>(out2, pred, (float*)d_out);
}
